// CausalHierarchicalMemoryLM_83399674954365
// MI455X (gfx1250) — matrix operand order repaired and checked by reading, compiled, not run on hardware
//
#include <hip/hip_runtime.h>

// ---------------- problem constants (from reference) ----------------
static constexpr int kV = 32000;
static constexpr int kD = 512;
static constexpr int kR = 64;
static constexpr int kTopK = 16;
static constexpr int kN0 = 256, kN1 = 64, kN2 = 16;
static constexpr int kChunk = 64;
static constexpr int kB = 2;
static constexpr int kS = 2048;
static constexpr int kNch = kS / kChunk;      // 32
static constexpr float kScale = 0.125f;       // 1/sqrt(64)
static constexpr float kNeg = -1e30f;

// ---------------- helpers ----------------
__device__ inline unsigned short f2bf(float f) {
  unsigned u = __float_as_uint(f);
  unsigned r = u + 0x7FFFu + ((u >> 16) & 1u);  // round-to-nearest-even
  return (unsigned short)(r >> 16);
}

// ---------------- tiny kernels ----------------
__global__ void k_zero(float* __restrict__ p, int n) {
  int i = blockIdx.x * blockDim.x + threadIdx.x;
  if (i < n) p[i] = 0.0f;
}

__global__ void k_embed(const int* __restrict__ ids, const float* __restrict__ emb,
                        float* __restrict__ x) {
  int row = blockIdx.x;                       // b*S + s
  int id = ids[row];
  const float* e = emb + (size_t)id * kD;
  float* xr = x + (size_t)row * kD;
  for (int d = threadIdx.x; d < kD; d += blockDim.x) xr[d] = e[d];
}

__global__ void k_cvt_bf16(const float* __restrict__ src, unsigned short* __restrict__ dst, int n) {
  int i = blockIdx.x * blockDim.x + threadIdx.x;
  if (i < n) dst[i] = f2bf(src[i]);
}

// ---------------- sequential scan (one workgroup per batch) ----------------
__device__ void dev_proj(const float* __restrict__ src, int M,
                         const float* __restrict__ w,   // [D,R]
                         float* __restrict__ outb, int tid, int nthr) {
  for (int i = tid; i < M * kR; i += nthr) {
    int m = i >> 6, r = i & 63;
    const float* srow = src + (size_t)m * kD;
    const float* wc = w + r;
    float acc = 0.0f;
    for (int d = 0; d < kD; ++d) acc += srow[d] * wc[(size_t)d * kR];
    outb[i] = acc;
  }
}

__device__ void dev_scores(const float* __restrict__ Ab, int M,
                           const float* __restrict__ Bb, int N,
                           float* __restrict__ sb, int tid, int nthr) {
  for (int i = tid; i < M * N; i += nthr) {
    int m = i / N, n = i % N;
    const float* ar = Ab + (size_t)m * kR;
    const float* br = Bb + (size_t)n * kR;
    float acc = 0.0f;
    for (int r = 0; r < kR; ++r) acc += ar[r] * br[r];
    sb[i] = acc * kScale;
  }
}

// top-16 per row + softmax; signed_mode==1: sign(v)*softmax(|v|)*st[idx]
__device__ void dev_topk(float* __restrict__ sb, int M, int N,
                         float* __restrict__ wbv, int* __restrict__ tbv,
                         const float* __restrict__ st, int signed_mode, int tid) {
  if (tid < M) {
    float* row = sb + (size_t)tid * N;
    float tv[kTopK]; int ti[kTopK];
    for (int k = 0; k < kTopK; ++k) {
      float best = -3.0e38f; int bi = 0;
      for (int j = 0; j < N; ++j) { float v = row[j]; if (v > best) { best = v; bi = j; } }
      tv[k] = best; ti[k] = bi; row[bi] = -3.0e38f;
    }
    float e[kTopK]; float sum = 0.0f;
    if (!signed_mode) {
      float mx = tv[0];
      for (int k = 0; k < kTopK; ++k) { e[k] = __expf(tv[k] - mx); sum += e[k]; }
      float inv = 1.0f / sum;
      for (int k = 0; k < kTopK; ++k) { wbv[tid * kTopK + k] = e[k] * inv; tbv[tid * kTopK + k] = ti[k]; }
    } else {
      float mx = 0.0f;
      for (int k = 0; k < kTopK; ++k) mx = fmaxf(mx, fabsf(tv[k]));
      for (int k = 0; k < kTopK; ++k) { e[k] = __expf(fabsf(tv[k]) - mx); sum += e[k]; }
      float inv = 1.0f / sum;
      for (int k = 0; k < kTopK; ++k) {
        float sg = (tv[k] > 0.0f) ? 1.0f : ((tv[k] < 0.0f) ? -1.0f : 0.0f);
        wbv[tid * kTopK + k] = sg * e[k] * inv * st[ti[k]];
        tbv[tid * kTopK + k] = ti[k];
      }
    }
  }
}

__device__ void dev_scatter(float* __restrict__ mem, const float* __restrict__ src, int M,
                            const float* __restrict__ wbv, const int* __restrict__ tbv,
                            int tid, int nthr) {
  for (int p = tid; p < M * kTopK; p += nthr) {
    int m = p / kTopK;
    float w = wbv[p]; int idx = tbv[p];
    float* mr = mem + (size_t)idx * kD;
    const float* sr = src + (size_t)m * kD;
    for (int d = 0; d < kD; ++d) atomicAdd(&mr[d], w * sr[d]);
  }
}

__device__ void dev_ln_state(const float* __restrict__ mem, int N,
                             const float* __restrict__ w_state, float b0,
                             float* __restrict__ h, float* __restrict__ st, int tid) {
  if (tid < N) {
    const float* row = mem + (size_t)tid * kD;
    float mu = 0.0f;
    for (int d = 0; d < kD; ++d) mu += row[d];
    mu *= (1.0f / kD);
    float var = 0.0f;
    for (int d = 0; d < kD; ++d) { float t = row[d] - mu; var += t * t; }
    var *= (1.0f / kD);
    float rs = rsqrtf(var + 1e-5f);
    float dot = 0.0f;
    float* hr = h + (size_t)tid * kD;
    for (int d = 0; d < kD; ++d) { hr[d] = (row[d] - mu) * rs; dot += row[d] * w_state[d]; }
    st[tid] = dot + b0;
  }
}

__device__ void dev_gather_delta(const float* __restrict__ mem, int N,
                                 const float* __restrict__ wbv, const int* __restrict__ tbv,
                                 float* __restrict__ db, int tid, int nthr) {
  for (int i = tid; i < N * kD; i += nthr) {
    int n = i / kD, d = i % kD;
    const float* wr = wbv + n * kTopK;
    const int* tr = tbv + n * kTopK;
    float acc = 0.0f;
    for (int k = 0; k < kTopK; ++k) acc += wr[k] * mem[(size_t)tr[k] * kD + d];
    db[i] = acc;
  }
}

__device__ void dev_route_write(const float* src, int M, float* mem, int N,
                                const float* rq, const float* rk,
                                float* Ab, float* Bb, float* sb, float* wbv, int* tbv,
                                int tid, int nthr) {
  dev_proj(src, M, rq, Ab, tid, nthr);
  dev_proj(mem, N, rk, Bb, tid, nthr);
  __syncthreads();
  dev_scores(Ab, M, Bb, N, sb, tid, nthr);
  __syncthreads();
  dev_topk(sb, M, N, wbv, tbv, nullptr, 0, tid);
  __syncthreads();
  dev_scatter(mem, src, M, wbv, tbv, tid, nthr);
  __syncthreads();
}

__device__ void dev_propagate(float* mem, int N, const float* pa, const float* pb,
                              const float* w_state, float b0,
                              float* hb, float* db, float* Ab, float* Bb, float* sb,
                              float* wbv, int* tbv, float* st, int tid, int nthr) {
  dev_ln_state(mem, N, w_state, b0, hb, st, tid);
  __syncthreads();
  dev_proj(hb, N, pa, Ab, tid, nthr);
  dev_proj(hb, N, pb, Bb, tid, nthr);
  __syncthreads();
  dev_scores(Ab, N, Bb, N, sb, tid, nthr);
  __syncthreads();
  dev_topk(sb, N, N, wbv, tbv, st, 1, tid);
  __syncthreads();
  dev_gather_delta(mem, N, wbv, tbv, db, tid, nthr);
  __syncthreads();
  for (int i = tid; i < N * kD; i += nthr) mem[i] += db[i];
  __syncthreads();
}

__global__ __launch_bounds__(1024) void k_scan(
    const float* __restrict__ x, const float* __restrict__ w_state,
    const float* __restrict__ b_state, const float* __restrict__ w_pred,
    const float* __restrict__ route_q, const float* __restrict__ route_k,
    const float* __restrict__ prop_a, const float* __restrict__ prop_b,
    const float* __restrict__ read_q, const float* __restrict__ read_k,
    float* v0, float* v1, float* v2,
    float* hbuf, float* dbuf, float* Abuf, float* Bbuf, float* sbuf,
    float* wbuf, int* tibuf, float* stbuf,
    float* __restrict__ qarr) {
  const int b = blockIdx.x, tid = threadIdx.x, nthr = blockDim.x;
  const float b0 = b_state[0];
  float* V0 = v0 + (size_t)b * kN0 * kD;
  float* V1 = v1 + (size_t)b * kN1 * kD;
  float* V2 = v2 + (size_t)b * kN2 * kD;
  float* hb = hbuf + (size_t)b * kN0 * kD;
  float* db = dbuf + (size_t)b * kN0 * kD;
  float* Ab = Abuf + (size_t)b * kN0 * kR;
  float* Bb = Bbuf + (size_t)b * kN0 * kR;
  float* sb = sbuf + (size_t)b * kN0 * kN0;
  float* wb = wbuf + (size_t)b * kN0 * kTopK;
  int* tb = tibuf + (size_t)b * kN0 * kTopK;
  float* stb = stbuf + (size_t)b * kN0;

  for (int t = 0; t < kNch; ++t) {
    const float* xc = x + ((size_t)b * kS + (size_t)t * kChunk) * kD;
    // level 0
    dev_route_write(xc, kChunk, V0, kN0, route_q + 0 * kD * kR, route_k + 0 * kD * kR,
                    Ab, Bb, sb, wb, tb, tid, nthr);
    dev_propagate(V0, kN0, prop_a + 0 * kD * kR, prop_b + 0 * kD * kR, w_state, b0,
                  hb, db, Ab, Bb, sb, wb, tb, stb, tid, nthr);
    // level 1 (src = v0)
    dev_route_write(V0, kN0, V1, kN1, route_q + 1 * kD * kR, route_k + 1 * kD * kR,
                    Ab, Bb, sb, wb, tb, tid, nthr);
    dev_propagate(V1, kN1, prop_a + 1 * kD * kR, prop_b + 1 * kD * kR, w_state, b0,
                  hb, db, Ab, Bb, sb, wb, tb, stb, tid, nthr);
    // level 2 (src = v1)
    dev_route_write(V1, kN1, V2, kN2, route_q + 2 * kD * kR, route_k + 2 * kD * kR,
                    Ab, Bb, sb, wb, tb, tid, nthr);
    dev_propagate(V2, kN2, prop_a + 2 * kD * kR, prop_b + 2 * kD * kR, w_state, b0,
                  hb, db, Ab, Bb, sb, wb, tb, stb, tid, nthr);
    // read from v0 + w_pred projection
    dev_proj(xc, kChunk, read_q, Ab, tid, nthr);
    dev_proj(V0, kN0, read_k, Bb, tid, nthr);
    __syncthreads();
    dev_scores(Ab, kChunk, Bb, kN0, sb, tid, nthr);
    __syncthreads();
    dev_topk(sb, kChunk, kN0, wb, tb, nullptr, 0, tid);
    __syncthreads();
    float* qout = qarr + ((size_t)b * kS + (size_t)t * kChunk) * kD;
    for (int i = tid; i < kChunk * kD; i += nthr) {
      int m = i / kD, d = i % kD;
      const float* xr = xc + (size_t)m * kD;
      float acc = 0.0f;
      for (int dd = 0; dd < kD; ++dd) acc += xr[dd] * w_pred[(size_t)dd * kD + d];
      float rd = 0.0f;
      for (int k = 0; k < kTopK; ++k)
        rd += wb[m * kTopK + k] * V0[(size_t)tb[m * kTopK + k] * kD + d];
      qout[i] = acc + rd;
    }
    __syncthreads();
  }
}

// ---------------- prediction head ----------------
__global__ void k_ln_state(const float* __restrict__ q, const float* __restrict__ w_state,
                           const float* __restrict__ b_state,
                           float* __restrict__ h, float* __restrict__ st) {
  int r = blockIdx.x * blockDim.x + threadIdx.x;
  if (r >= kB * kS) return;
  const float* row = q + (size_t)r * kD;
  float mu = 0.0f;
  for (int d = 0; d < kD; ++d) mu += row[d];
  mu *= (1.0f / kD);
  float var = 0.0f;
  for (int d = 0; d < kD; ++d) { float t = row[d] - mu; var += t * t; }
  var *= (1.0f / kD);
  float rs = rsqrtf(var + 1e-5f);
  float dot = 0.0f;
  float* hr = h + (size_t)r * kD;
  for (int d = 0; d < kD; ++d) { hr[d] = (row[d] - mu) * rs; dot += row[d] * w_state[d]; }
  st[r] = dot + b_state[0];
}

__global__ void k_proj2(const float* __restrict__ h, const float* __restrict__ wa,
                        const float* __restrict__ wb, float* __restrict__ a,
                        float* __restrict__ bo) {
  int i = blockIdx.x * blockDim.x + threadIdx.x;
  if (i >= kB * kS * kR) return;
  int row = i / kR, r = i % kR;
  const float* hr = h + (size_t)row * kD;
  float s1 = 0.0f, s2 = 0.0f;
  for (int d = 0; d < kD; ++d) {
    float v = hr[d];
    s1 += v * wa[(size_t)d * kR + r];
    s2 += v * wb[(size_t)d * kR + r];
  }
  a[i] = s1; bo[i] = s2;
}

__global__ __launch_bounds__(256) void k_window(const float* __restrict__ h,
                                                const float* __restrict__ aab,
                                                const float* __restrict__ bbb,
                                                const float* __restrict__ stq,
                                                float* __restrict__ q) {
  int gw = (blockIdx.x * blockDim.x + threadIdx.x) >> 5;  // one wave per position
  int lane = threadIdx.x & 31;
  if (gw >= kB * kS) return;
  int bi = gw / kS, s = gw % kS;
  const float* arow = aab + (size_t)gw * kR;
  int i0 = s - lane, i1 = s - lane - 32;
  int c0 = i0 < 0 ? 0 : i0, c1 = i1 < 0 ? 0 : i1;
  bool m0 = i0 >= 0, m1 = i1 >= 0;
  const float* br0 = bbb + ((size_t)bi * kS + c0) * kR;
  const float* br1 = bbb + ((size_t)bi * kS + c1) * kR;
  float sc0 = 0.0f, sc1 = 0.0f;
  for (int r = 0; r < kR; ++r) { float av = arow[r]; sc0 += av * br0[r]; sc1 += av * br1[r]; }
  sc0 *= kScale; sc1 *= kScale;
  float ab0 = m0 ? fabsf(sc0) : kNeg;
  float ab1 = m1 ? fabsf(sc1) : kNeg;
  float mx = fmaxf(ab0, ab1);
  for (int o = 16; o > 0; o >>= 1) mx = fmaxf(mx, __shfl_xor(mx, o, 32));
  float e0 = __expf(ab0 - mx), e1 = __expf(ab1 - mx);
  float sum = e0 + e1;
  for (int o = 16; o > 0; o >>= 1) sum += __shfl_xor(sum, o, 32);
  float inv = 1.0f / sum;
  float sg0 = (sc0 > 0.0f) ? 1.0f : ((sc0 < 0.0f) ? -1.0f : 0.0f);
  float sg1 = (sc1 > 0.0f) ? 1.0f : ((sc1 < 0.0f) ? -1.0f : 0.0f);
  float we0 = m0 ? sg0 * e0 * inv * stq[(size_t)bi * kS + c0] : 0.0f;
  float we1 = m1 ? sg1 * e1 * inv * stq[(size_t)bi * kS + c1] : 0.0f;
  float acc[16];
#pragma unroll
  for (int j = 0; j < 16; ++j) acc[j] = 0.0f;
  for (int w = 0; w < 64; ++w) {
    float wew = __shfl((w < 32) ? we0 : we1, w & 31, 32);
    if (wew != 0.0f) {                       // wave-uniform branch
      int iw = s - w; int cw = iw < 0 ? 0 : iw;
      const float* hr = h + ((size_t)bi * kS + cw) * kD;
#pragma unroll
      for (int j = 0; j < 16; ++j) acc[j] += wew * hr[lane + j * 32];
    }
  }
  float* qr = q + ((size_t)bi * kS + s) * kD;
#pragma unroll
  for (int j = 0; j < 16; ++j) qr[lane + j * 32] += acc[j];
}

__global__ void k_ln_bf16(const float* __restrict__ q, unsigned short* __restrict__ hb) {
  int r = blockIdx.x * blockDim.x + threadIdx.x;
  if (r >= kB * kS) return;
  const float* row = q + (size_t)r * kD;
  float mu = 0.0f;
  for (int d = 0; d < kD; ++d) mu += row[d];
  mu *= (1.0f / kD);
  float var = 0.0f;
  for (int d = 0; d < kD; ++d) { float t = row[d] - mu; var += t * t; }
  var *= (1.0f / kD);
  float rs = rsqrtf(var + 1e-5f);
  unsigned short* hr = hb + (size_t)r * kD;
  for (int d = 0; d < kD; ++d) hr[d] = f2bf((row[d] - mu) * rs);
}

// ---------------- lm_head GEMM: [4096,512] x [512,32000] via bf16 WMMA ----------------
// Workgroup tile: 128 (M) x 64 (N). 8 waves; each wave owns 16 rows x 64 cols
// (4 v8f accumulators, one A fragment reused across 4 WMMAs per K step).
// B panel (emb rows for this N tile) is staged in LDS with
// global_load_async_to_lds_b128 (ASYNCcnt), double-buffered in K chunks of 128,
// overlapped with WMMA via s_wait_asynccnt 0x4. LDS rows padded 16B to de-skew banks.
typedef __attribute__((ext_vector_type(16))) __bf16 v16bf;
typedef __attribute__((ext_vector_type(8))) float v8f;
union FragBF { v16bf v; uint4 q[2]; };

static constexpr int kKCH = 128;                 // K chunk staged in LDS
static constexpr int kBPad = kKCH + 8;           // +16B row pad (keeps 16B alignment)

__global__ __launch_bounds__(256) void k_lmhead(const unsigned short* __restrict__ Abf,
                                                const unsigned short* __restrict__ Ebf,
                                                float* __restrict__ out) {
  __shared__ unsigned short tileB[2][64 * kBPad];   // 2 x ~17KB
  const int nBlocksN = kV / 64;                     // 500
  const int m0 = (blockIdx.x / nBlocksN) * 128;
  const int n0 = (blockIdx.x % nBlocksN) * 64;
  const int tid = threadIdx.x;
  const int wv = tid >> 5, lane = tid & 31;
  const int hl = lane & 15;
  const bool hi = lane >= 16;

  // low 32 bits of generic pointer == wave-relative LDS byte offset (ISA 10.2)
  unsigned ldsb0 = (unsigned)(size_t)(void*)&tileB[0][0];
  unsigned ldsb1 = (unsigned)(size_t)(void*)&tileB[1][0];

  // stage K chunk kc into LDS buffer buf: 64 cols x 128 K of bf16 = 1024 x b128
  auto stage = [&](int kc, unsigned ldsbase) {
    int base = tid * 4;
#pragma unroll
    for (int jj = 0; jj < 4; ++jj) {
      int j = base + jj;
      int col = j >> 4;                        // 16 transfers per column
      int seg = j & 15;                        // 8 bf16 each
      const unsigned short* g = Ebf + (size_t)(n0 + col) * kD + kc * kKCH + seg * 8;
      unsigned loff = ldsbase + (unsigned)(col * kBPad + seg * 8) * 2u;
      asm volatile("global_load_async_to_lds_b128 %0, %1, off"
                   :: "v"(loff), "v"(g) : "memory");
    }
  };

  const unsigned short* abase = Abf + (size_t)(m0 + wv * 16 + hl) * kD + (hi ? 8 : 0);
  v8f acc[4];
  acc[0] = {}; acc[1] = {}; acc[2] = {}; acc[3] = {};

  stage(0, ldsb0);
  for (int kc = 0; kc < kD / kKCH; ++kc) {
    const int buf = kc & 1;
    if (kc + 1 < kD / kKCH) {
      stage(kc + 1, buf ? ldsb0 : ldsb1);                 // prefetch next chunk
      asm volatile("s_wait_asynccnt 0x4" ::: "memory");   // current chunk landed
    } else {
      asm volatile("s_wait_asynccnt 0x0" ::: "memory");
    }
    __syncthreads();
    const unsigned short* bbuf = &tileB[buf][0];
#pragma unroll
    for (int kk = 0; kk < kKCH; kk += 32) {
      const int k0 = kc * kKCH + kk;
      FragBF af;
      af.q[0] = *(const uint4*)(abase + k0);        // K+0..7   (pairs K=2v,2v+1)
      af.q[1] = *(const uint4*)(abase + k0 + 16);   // K+16..23
      __builtin_prefetch(abase + k0 + 32, 0, 0);    // global_prefetch_b8
#pragma unroll
      for (int j = 0; j < 4; ++j) {
        FragBF bf2;
        const uint4* bp = (const uint4*)(bbuf + (j * 16 + hl) * kBPad + kk + (hi ? 8 : 0));
        bf2.q[0] = bp[0];                           // ds_load_b128: K+0..7
        bf2.q[1] = bp[2];                           // K+16..23
        acc[j] = __builtin_amdgcn_wmma_f32_16x16x32_bf16(false, af.v, false, bf2.v,
                                                         (short)0, acc[j], false, false);
      }
    }
    __syncthreads();   // buf may be re-staged next iteration
  }

  // C/D layout: lane -> col hl; VGPR i -> row base+(hi?8:0)+i
  const int rbase = m0 + wv * 16 + (hi ? 8 : 0);
#pragma unroll
  for (int j = 0; j < 4; ++j) {
    float* orow = out + (size_t)rbase * kV + (size_t)(n0 + j * 16 + hl);
#pragma unroll
    for (int i = 0; i < 8; ++i) orow[(size_t)i * kV] = acc[j][i];
  }
}

// ---------------- launch ----------------
extern "C" void kernel_launch(void* const* d_in, const int* in_sizes, int n_in,
                              void* d_out, int out_size, void* d_ws, size_t ws_size,
                              hipStream_t stream) {
  const int* ids = (const int*)d_in[0];
  const float* emb = (const float*)d_in[1];
  const float* w_state = (const float*)d_in[2];
  const float* b_state = (const float*)d_in[3];
  const float* w_pred = (const float*)d_in[4];
  const float* route_q = (const float*)d_in[5];
  const float* route_k = (const float*)d_in[6];
  const float* prop_a = (const float*)d_in[7];
  const float* prop_b = (const float*)d_in[8];
  const float* read_q = (const float*)d_in[9];
  const float* read_k = (const float*)d_in[10];
  const float* pred_a = (const float*)d_in[11];
  const float* pred_b = (const float*)d_in[12];
  float* out = (float*)d_out;

  char* ws = (char*)d_ws;
  size_t off = 0;
  auto alloc = [&](size_t bytes) -> void* {
    void* p = ws + off;
    off += (bytes + 255) & ~(size_t)255;
    return p;
  };
  float* x    = (float*)alloc((size_t)kB * kS * kD * 4);
  float* q    = (float*)alloc((size_t)kB * kS * kD * 4);
  float* v0   = (float*)alloc((size_t)kB * kN0 * kD * 4);
  float* v1   = (float*)alloc((size_t)kB * kN1 * kD * 4);
  float* v2   = (float*)alloc((size_t)kB * kN2 * kD * 4);
  float* hbuf = (float*)alloc((size_t)kB * kN0 * kD * 4);
  float* dbuf = (float*)alloc((size_t)kB * kN0 * kD * 4);
  float* Abuf = (float*)alloc((size_t)kB * kN0 * kR * 4);
  float* Bbuf = (float*)alloc((size_t)kB * kN0 * kR * 4);
  float* sbuf = (float*)alloc((size_t)kB * kN0 * kN0 * 4);
  float* wbuf = (float*)alloc((size_t)kB * kN0 * kTopK * 4);
  int*  tibuf = (int*)alloc((size_t)kB * kN0 * kTopK * 4);
  float* stbuf = (float*)alloc((size_t)kB * kN0 * 4);
  float* hq   = (float*)alloc((size_t)kB * kS * kD * 4);
  float* abuf = (float*)alloc((size_t)kB * kS * kR * 4);
  float* bbuf = (float*)alloc((size_t)kB * kS * kR * 4);
  float* stq  = (float*)alloc((size_t)kB * kS * 4);
  unsigned short* hbb = (unsigned short*)alloc((size_t)kB * kS * kD * 2);
  unsigned short* ebb = (unsigned short*)alloc((size_t)kV * kD * 2);
  (void)ws_size; (void)in_sizes; (void)n_in; (void)out_size;

  // 1. embedding gather
  k_embed<<<kB * kS, 256, 0, stream>>>(ids, emb, x);
  // 2. zero memory states (deterministic per call)
  k_zero<<<(kB * kN0 * kD + 255) / 256, 256, 0, stream>>>(v0, kB * kN0 * kD);
  k_zero<<<(kB * kN1 * kD + 255) / 256, 256, 0, stream>>>(v1, kB * kN1 * kD);
  k_zero<<<(kB * kN2 * kD + 255) / 256, 256, 0, stream>>>(v2, kB * kN2 * kD);
  // 3. sequential hierarchical scan (one workgroup per batch)
  k_scan<<<kB, 1024, 0, stream>>>(x, w_state, b_state, w_pred, route_q, route_k,
                                  prop_a, prop_b, read_q, read_k,
                                  v0, v1, v2, hbuf, dbuf, Abuf, Bbuf, sbuf,
                                  wbuf, tibuf, stbuf, q);
  // 4. prediction head: two causal windowed sparse layers
  for (int l = 0; l < 2; ++l) {
    k_ln_state<<<(kB * kS + 255) / 256, 256, 0, stream>>>(q, w_state, b_state, hq, stq);
    k_proj2<<<(kB * kS * kR + 255) / 256, 256, 0, stream>>>(
        hq, pred_a + (size_t)l * kD * kR, pred_b + (size_t)l * kD * kR, abuf, bbuf);
    k_window<<<(kB * kS) / 8, 256, 0, stream>>>(hq, abuf, bbuf, stq, q);
  }
  // 5. final layernorm -> bf16; embedding -> bf16
  k_ln_bf16<<<(kB * kS + 255) / 256, 256, 0, stream>>>(q, hbb);
  k_cvt_bf16<<<(kV * kD + 255) / 256, 256, 0, stream>>>(emb, ebb, kV * kD);
  // 6. lm_head GEMM: LDS-staged, double-buffered async B, bf16 WMMA
  int nBlocks = (kB * kS / 128) * (kV / 64);        // 32 * 500 = 16000
  k_lmhead<<<nBlocks, 256, 0, stream>>>(hbb, ebb, out);
}
